// AttentionNet_13460427506017
// MI455X (gfx1250) — matrix operand order repaired and checked by reading, not compiled, not run on hardware
//
#include <hip/hip_runtime.h>
#include <hip/hip_bf16.h>
#include <math.h>

typedef __attribute__((ext_vector_type(16))) __bf16 bf16x16;
typedef __attribute__((ext_vector_type(8)))  __bf16 bf16x8;
typedef __attribute__((ext_vector_type(8)))  float  f32x8;

#define IMGSZ 448
#define PADSZ 224
#define PARTSZ 224
#define NANCH 1614
#define TOPN 4
#define NCLASS 200
#define NBATCH 8

#define CEILDIV(a,b) (((a)+(b)-1)/(b))

// ---------------------------------------------------------------- f32 -> bf16
__global__ void cvt_bf16_kernel(const float* __restrict__ src,
                                __bf16* __restrict__ dst, long long n) {
  long long i = (long long)blockIdx.x * blockDim.x + threadIdx.x;
  if (i < n) dst[i] = (__bf16)src[i];
}

// ---------------------------------------------------------------- im2col (bf16 out)
// dst is K x Nld row-major, K = C*KH*KW, column n = b*OH*OW + oy*OW + ox.
// srcNld == 0  : src is NCHW f32
// srcNld != 0  : src is [C][srcNld] with column = b*H*W + iy*W + ix (GEMM-out layout)
__global__ void im2col_kernel(const float* __restrict__ src, __bf16* __restrict__ dst,
                              int B, int C, int H, int W, int KH, int KW,
                              int stride, int pad, int OH, int OW,
                              int Nld, int srcNld) {
  long long K = (long long)C * KH * KW;
  long long total = K * Nld;
  long long idx = (long long)blockIdx.x * blockDim.x + threadIdx.x;
  if (idx >= total) return;
  int n = (int)(idx % Nld);
  int k = (int)(idx / Nld);
  int Nvalid = B * OH * OW;
  float v = 0.0f;
  if (n < Nvalid) {
    int b = n / (OH * OW);
    int pos = n % (OH * OW);
    int oy = pos / OW, ox = pos % OW;
    int kx = k % KW;
    int ky = (k / KW) % KH;
    int c  = k / (KW * KH);
    int iy = oy * stride - pad + ky;
    int ix = ox * stride - pad + kx;
    if (iy >= 0 && iy < H && ix >= 0 && ix < W) {
      long long o;
      if (srcNld == 0) o = (((long long)b * C + c) * H + iy) * W + ix;
      else             o = (long long)c * srcNld + (long long)b * H * W
                           + (long long)iy * W + ix;
      v = src[o];
    }
  }
  dst[idx] = (__bf16)v;
}

// ---------------------------------------------------------------- WMMA GEMM
__device__ __forceinline__ bf16x16 ld_frag16(const __bf16* p0, const __bf16* p1) {
  bf16x8 lo = *(const bf16x8*)p0;
  bf16x8 hi = *(const bf16x8*)p1;
  bf16x16 r;
#pragma unroll
  for (int i = 0; i < 8; ++i) { r[i] = lo[i]; r[i + 8] = hi[i]; }
  return r;
}

// C[M x Nld] f32 = A[M x K] bf16 (row-major) * B[K x Nld] bf16 (row-major)
// M % 128 == 0, Nld % 64 == 0, K % 64 == 0.
// 8 waves/block stacked in M (block tile 128x64); each wave owns a 16x64 strip.
// 2-deep software pipeline, unrolled x2: each half-iteration's loads are issued
// before a 4-WMMA chain that does not consume them, so the matrix pipe hides
// the global-load latency, and the two stages use disjoint register sets
// (no cross-iteration register rotation).
__global__ void __launch_bounds__(256)
gemm_wmma_bf16(const __bf16* __restrict__ A, const __bf16* __restrict__ Bm,
               float* __restrict__ C, const float* __restrict__ bias,
               int M, int Nld, int K, int relu) {
  int lane = threadIdx.x & 31;
  int wave = threadIdx.x >> 5;
  int tileM = blockIdx.x * 128 + wave * 16;
  int tileN = blockIdx.y * 64;
  int half = lane >> 4;       // 0: lanes 0-15, 1: lanes 16-31
  int l16  = lane & 15;

  // A fragment: row = tileM + l16; K slots {kb..kb+7, kb+16..kb+23}, kb = half*8
  const __bf16* arow = A + (size_t)(tileM + l16) * K + half * 8;
  // B fragments: column = tileN + t*16 + l16; K slots {kb..kb+7, kb+16..kb+23}, kb = half*8
  const __bf16* brow = Bm + (size_t)(half * 8) * Nld + tileN + l16;
  const size_t bstep32 = (size_t)Nld * 32;   // B element offset per 32-K step

  f32x8 acc[4];
#pragma unroll
  for (int t = 0; t < 4; ++t)
#pragma unroll
    for (int r = 0; r < 8; ++r) acc[t][r] = 0.0f;

  bf16x16 a0 = ld_frag16(arow, arow + 16);
  bf16x16 b0[4];
#pragma unroll
  for (int t = 0; t < 4; ++t) for (int s = 0; s < 16; ++s) b0[t][s] = brow[(size_t)(s + (s & 8)) * Nld + t * 16];

  size_t bOff = 0;                 // B offset for current k0
  for (int k0 = 0; k0 < K; k0 += 64) {
    // ---- stage-1 loads (k0+32); always in range since K % 64 == 0
    const __bf16* ap1 = arow + k0 + 32;
    const __bf16* bp1 = brow + bOff + bstep32;
    bf16x16 a1 = ld_frag16(ap1, ap1 + 16);
    bf16x16 b1[4];
#pragma unroll
    for (int t = 0; t < 4; ++t) for (int s = 0; s < 16; ++s) b1[t][s] = bp1[(size_t)(s + (s & 8)) * Nld + t * 16];
    __builtin_prefetch((const void*)(ap1 + 96), 0, 3);
    __builtin_prefetch((const void*)(bp1 + 3 * bstep32), 0, 3);

    // ---- 4 WMMAs on stage-0 fragments (overlaps the loads above)
#pragma unroll
    for (int t = 0; t < 4; ++t)
      acc[t] = __builtin_amdgcn_wmma_f32_16x16x32_bf16(
          false, a0, false, b0[t], (short)0, acc[t], false, false);

    // ---- stage-2 loads (k0+64); wrap on the final iteration (result unused)
    int k2 = (k0 + 64 < K) ? (k0 + 64) : k0;
    size_t bOff2 = (k0 + 64 < K) ? (bOff + 2 * bstep32) : bOff;
    const __bf16* ap2 = arow + k2;
    const __bf16* bp2 = brow + bOff2;
    a0 = ld_frag16(ap2, ap2 + 16);
#pragma unroll
    for (int t = 0; t < 4; ++t) for (int s = 0; s < 16; ++s) b0[t][s] = bp2[(size_t)(s + (s & 8)) * Nld + t * 16];

    // ---- 4 WMMAs on stage-1 fragments (overlaps the loads above)
#pragma unroll
    for (int t = 0; t < 4; ++t)
      acc[t] = __builtin_amdgcn_wmma_f32_16x16x32_bf16(
          false, a1, false, b1[t], (short)0, acc[t], false, false);

    bOff = bOff2;
  }

#pragma unroll
  for (int t = 0; t < 4; ++t) {
#pragma unroll
    for (int r = 0; r < 8; ++r) {
      int m = tileM + r + half * 8;
      int n = tileN + t * 16 + l16;
      float v = acc[t][r];
      if (bias) v += bias[m];
      if (relu) v = fmaxf(v, 0.0f);
      C[(size_t)m * Nld + n] = v;
    }
  }
}

// ---------------------------------------------------------------- mean pool
// src [Cch][Nld] (col = b*per + p)  ->  dst [B][Cch]
__global__ void mean_pool_kernel(const float* __restrict__ src, float* __restrict__ dst,
                                 int Cch, int Nld, int per, int B) {
  int i = blockIdx.x * blockDim.x + threadIdx.x;
  if (i >= B * Cch) return;
  int b = i / Cch, c = i % Cch;
  const float* p = src + (size_t)c * Nld + (size_t)b * per;
  float s = 0.f;
  for (int j = 0; j < per; ++j) s += p[j];
  dst[(size_t)b * Cch + c] = s / (float)per;
}

// ---------------------------------------------------------------- dense FC
__global__ void fc_kernel(const float* __restrict__ feat, const float* __restrict__ W,
                          const float* __restrict__ bias, float* __restrict__ out,
                          int B, int M, int K) {
  int i = blockIdx.x * blockDim.x + threadIdx.x;
  if (i >= B * M) return;
  int b = i / M, m = i % M;
  const float* f = feat + (size_t)b * K;
  const float* w = W + (size_t)m * K;
  float s = bias[m];
  for (int k = 0; k < K; ++k) s += f[k] * w[k];
  out[i] = s;
}

// ---------------------------------------------------------------- concat FC
__global__ void concat_fc_kernel(const float* __restrict__ part_feats,
                                 const float* __restrict__ feature,
                                 const float* __restrict__ W,
                                 const float* __restrict__ bias,
                                 float* __restrict__ out, int B, int M) {
  int i = blockIdx.x * blockDim.x + threadIdx.x;
  if (i >= B * M) return;
  int b = i / M, m = i % M;
  const float* w = W + (size_t)m * 10240;
  float s = bias[m];
  const float* pf = part_feats + (size_t)b * 8192;   // 4 parts x 2048, contiguous
  for (int j = 0; j < 8192; ++j) s += w[j] * pf[j];
  const float* ft = feature + (size_t)b * 2048;
  for (int j = 0; j < 2048; ++j) s += w[8192 + j] * ft[j];
  out[(size_t)b * M + m] = s;
}

// ---------------------------------------------------------------- 1x1 t-convs
// src [128][Nld], writes rpn[b*NANCH + off + c*per + pos]
__global__ void tconv_kernel(const float* __restrict__ src, const float* __restrict__ W,
                             const float* __restrict__ bias, float* __restrict__ rpn,
                             int Cout, int per, int Nld, int B, int off) {
  int i = blockIdx.x * blockDim.x + threadIdx.x;
  int total = B * Cout * per;
  if (i >= total) return;
  int b = i / (Cout * per);
  int rem = i % (Cout * per);
  int c = rem / per, pos = rem % per;
  float s = bias[c];
  for (int ch = 0; ch < 128; ++ch)
    s += W[c * 128 + ch] * src[(size_t)ch * Nld + (size_t)b * per + pos];
  rpn[(size_t)b * NANCH + off + c * per + pos] = s;
}

// ---------------------------------------------------------------- anchors
__device__ __forceinline__ void make_anchor(int i, int& y0, int& x0, int& y1, int& x1) {
  const float scl2[2] = {1.2599210498948732f, 1.5874010519681994f};
  const float scl3[3] = {1.0f, 1.2599210498948732f, 1.5874010519681994f};
  const float asp[3]  = {0.667f, 1.0f, 1.5f};
  int stride, fm, sa, pos; float size, s; int a;
  if (i < 1176)      { stride = 32;  fm = 14; size = 48.f;
                       sa = i / 196;            pos = i % 196; s = scl2[sa / 3]; a = sa % 3; }
  else if (i < 1470) { int j = i - 1176; stride = 64;  fm = 7;  size = 96.f;
                       sa = j / 49;             pos = j % 49;  s = scl2[sa / 3]; a = sa % 3; }
  else               { int j = i - 1470; stride = 128; fm = 4;  size = 192.f;
                       sa = j / 16;             pos = j % 16;  s = scl3[sa / 3]; a = sa % 3; }
  int iy = pos / fm, ix = pos % fm;
  float ar = asp[a];
  float h = size * s / sqrtf(ar);
  float w = size * s * sqrtf(ar);
  float cy = (iy + 0.5f) * (float)stride;
  float cx = (ix + 0.5f) * (float)stride;
  float fy0 = fminf(fmaxf(cy - 0.5f * h, 0.f), (float)IMGSZ) + (float)PADSZ;
  float fx0 = fminf(fmaxf(cx - 0.5f * w, 0.f), (float)IMGSZ) + (float)PADSZ;
  float fy1 = fminf(fmaxf(cy + 0.5f * h, 0.f), (float)IMGSZ) + (float)PADSZ;
  float fx1 = fminf(fmaxf(cx + 0.5f * w, 0.f), (float)IMGSZ) + (float)PADSZ;
  y0 = (int)fy0; x0 = (int)fx0; y1 = (int)fy1; x1 = (int)fx1;
}

// ---------------------------------------------------------------- hard NMS
__global__ void __launch_bounds__(256)
nms_kernel(const float* __restrict__ rpn, int* __restrict__ boxes_out,
           float* __restrict__ out, int idx_off, int prob_off) {
  __shared__ int   by0[NANCH], bx0[NANCH], by1[NANCH], bx1[NANCH];
  __shared__ float sc[NANCH];
  __shared__ float rv[256];
  __shared__ int   ri[256];
  int b = blockIdx.x;
  int t = threadIdx.x;
  for (int i = t; i < NANCH; i += 256) {
    int a0, a1, a2, a3;
    make_anchor(i, a0, a1, a2, a3);
    by0[i] = a0; bx0[i] = a1; by1[i] = a2; bx1[i] = a3;
    sc[i] = rpn[(size_t)b * NANCH + i];
  }
  __syncthreads();
  for (int step = 0; step < TOPN; ++step) {
    float best = -3.4e38f; int bi = 0;
    for (int i = t; i < NANCH; i += 256)
      if (sc[i] > best) { best = sc[i]; bi = i; }
    rv[t] = best; ri[t] = bi;
    __syncthreads();
    for (int sft = 128; sft > 0; sft >>= 1) {
      if (t < sft) {
        if (rv[t + sft] > rv[t] ||
            (rv[t + sft] == rv[t] && ri[t + sft] < ri[t])) {
          rv[t] = rv[t + sft]; ri[t] = ri[t + sft];
        }
      }
      __syncthreads();
    }
    int sel = ri[0];
    if (t == 0) {
      out[idx_off  + b * TOPN + step] = (float)sel;
      out[prob_off + b * TOPN + step] = rpn[(size_t)b * NANCH + sel];
      int* bo = boxes_out + (b * TOPN + step) * 4;
      bo[0] = by0[sel]; bo[1] = bx0[sel]; bo[2] = by1[sel]; bo[3] = bx1[sel];
    }
    __syncthreads();
    float sy0 = (float)by0[sel], sx0 = (float)bx0[sel];
    float sy1 = (float)by1[sel], sx1 = (float)bx1[sel];
    float sarea = (sy1 - sy0) * (sx1 - sx0);
    for (int i = t; i < NANCH; i += 256) {
      float y0 = (float)by0[i], x0 = (float)bx0[i];
      float y1 = (float)by1[i], x1 = (float)bx1[i];
      float area = (y1 - y0) * (x1 - x0);
      float iy = fmaxf(fminf(sy1, y1) - fmaxf(sy0, y0), 0.f);
      float ix = fmaxf(fminf(sx1, x1) - fmaxf(sx0, x0), 0.f);
      float inter = iy * ix;
      float iou = inter / (sarea + area - inter + 1e-9f);
      if (iou > 0.25f) sc[i] = -3.4e38f;
    }
    __syncthreads();
  }
}

// ---------------------------------------------------------------- bilinear crop
__global__ void crop_kernel(const float* __restrict__ x, const int* __restrict__ boxes,
                            float* __restrict__ parts) {
  int idx = blockIdx.x * blockDim.x + threadIdx.x;
  const int total = 32 * 3 * PARTSZ * PARTSZ;
  if (idx >= total) return;
  int px = idx % PARTSZ;
  int py = (idx / PARTSZ) % PARTSZ;
  int c  = (idx / (PARTSZ * PARTSZ)) % 3;
  int p  = idx / (3 * PARTSZ * PARTSZ);
  int b  = p / TOPN;
  const int* bo = boxes + p * 4;
  float y0 = (float)bo[0], x0 = (float)bo[1], y1 = (float)bo[2], x1 = (float)bo[3];
  float ty = y0 + py * ((y1 - y0 - 1.f) / (float)(PARTSZ - 1));
  float tx = x0 + px * ((x1 - x0 - 1.f) / (float)(PARTSZ - 1));
  const int Hp = IMGSZ + 2 * PADSZ;
  int iy0 = min(max((int)floorf(ty), 0), Hp - 1);
  int iy1 = min(iy0 + 1, Hp - 1);
  int ix0 = min(max((int)floorf(tx), 0), Hp - 1);
  int ix1 = min(ix0 + 1, Hp - 1);
  float wy = ty - (float)iy0, wx = tx - (float)ix0;
  auto sample = [&](int yy, int xx) -> float {
    yy -= PADSZ; xx -= PADSZ;
    if (yy < 0 || yy >= IMGSZ || xx < 0 || xx >= IMGSZ) return 0.f;
    return x[(((size_t)b * 3 + c) * IMGSZ + yy) * IMGSZ + xx];
  };
  float top = (1.f - wx) * sample(iy0, ix0) + wx * sample(iy0, ix1);
  float bot = (1.f - wx) * sample(iy1, ix0) + wx * sample(iy1, ix1);
  parts[idx] = (1.f - wy) * top + wy * bot;
}

// ================================================================ launcher
extern "C" void kernel_launch(void* const* d_in, const int* in_sizes, int n_in,
                              void* d_out, int out_size, void* d_ws, size_t ws_size,
                              hipStream_t stream) {
  const float* x     = (const float*)d_in[0];
  const float* W_bb  = (const float*)d_in[1];
  const float* b_bb  = (const float*)d_in[2];
  const float* W_fc  = (const float*)d_in[3];
  const float* b_fc  = (const float*)d_in[4];
  const float* W_d1  = (const float*)d_in[5];
  const float* b_d1  = (const float*)d_in[6];
  const float* W_d2  = (const float*)d_in[7];
  const float* b_d2  = (const float*)d_in[8];
  const float* W_d3  = (const float*)d_in[9];
  const float* b_d3  = (const float*)d_in[10];
  const float* W_t1  = (const float*)d_in[11];
  const float* b_t1  = (const float*)d_in[12];
  const float* W_t2  = (const float*)d_in[13];
  const float* b_t2  = (const float*)d_in[14];
  const float* W_t3  = (const float*)d_in[15];
  const float* b_t3  = (const float*)d_in[16];
  const float* W_cat = (const float*)d_in[17];
  const float* b_cat = (const float*)d_in[18];
  const float* W_part= (const float*)d_in[19];
  const float* b_part= (const float*)d_in[20];
  float* out = (float*)d_out;
  char* ws = (char*)d_ws;

  size_t off = 0;
  auto alloc = [&](size_t bytes) -> void* {
    void* p = (void*)(ws + off);
    off += (bytes + 255) & ~(size_t)255;
    return p;
  };
  __bf16* wbb    = (__bf16*)alloc(2048ull * 3072 * 2);
  __bf16* wd1    = (__bf16*)alloc(128ull * 18432 * 2);
  __bf16* wd2    = (__bf16*)alloc(128ull * 1152 * 2);
  __bf16* wd3    = (__bf16*)alloc(128ull * 1152 * 2);
  __bf16* colbuf = (__bf16*)alloc(18432ull * 1600 * 2);   // reused by every im2col
  float* conv_out= (float*)alloc(2048ull * 1600 * 4);     // reused for part conv
  float* feature = (float*)alloc(8ull * 2048 * 4);
  float* d1o     = (float*)alloc(128ull * 1600 * 4);
  float* d2o     = (float*)alloc(128ull * 448 * 4);
  float* d3o     = (float*)alloc(128ull * 128 * 4);
  float* rpn     = (float*)alloc(8ull * NANCH * 4);
  int*   boxes   = (int*)alloc(8ull * TOPN * 4 * 4);
  float* parts   = (float*)alloc(32ull * 3 * PARTSZ * PARTSZ * 4);
  float* pfeats  = (float*)alloc(32ull * 2048 * 4);

  // 1) weights -> bf16
  {
    long long n;
    n = 2048LL*3072; cvt_bf16_kernel<<<CEILDIV(n,256),256,0,stream>>>(W_bb, wbb, n);
    n = 128LL*18432; cvt_bf16_kernel<<<CEILDIV(n,256),256,0,stream>>>(W_d1, wd1, n);
    n = 128LL*1152;  cvt_bf16_kernel<<<CEILDIV(n,256),256,0,stream>>>(W_d2, wd2, n);
    n = 128LL*1152;  cvt_bf16_kernel<<<CEILDIV(n,256),256,0,stream>>>(W_d3, wd3, n);
  }

  // 2) backbone conv on full image: im2col + WMMA GEMM (M=2048,K=3072,N=1568->1600)
  {
    long long tot = 3072LL * 1600;
    im2col_kernel<<<CEILDIV(tot,256),256,0,stream>>>(x, colbuf, 8,3,IMGSZ,IMGSZ,
                                                     32,32,32,0,14,14, 1600, 0);
    dim3 g(2048/128, 1600/64);
    gemm_wmma_bf16<<<g,256,0,stream>>>(wbb, colbuf, conv_out, b_bb, 2048, 1600, 3072, 1);
    mean_pool_kernel<<<CEILDIV(8*2048,256),256,0,stream>>>(conv_out, feature, 2048, 1600, 196, 8);
    fc_kernel<<<CEILDIV(8*NCLASS,256),256,0,stream>>>(feature, W_fc, b_fc, out, 8, NCLASS, 2048); // raw_logits
  }

  // 3) proposal net
  {
    long long tot = 18432LL * 1600;
    im2col_kernel<<<CEILDIV(tot,256),256,0,stream>>>(conv_out, colbuf, 8,2048,14,14,
                                                     3,3,1,1,14,14, 1600, 1600);
    dim3 g1(128/128, 1600/64);
    gemm_wmma_bf16<<<g1,256,0,stream>>>(wd1, colbuf, d1o, b_d1, 128, 1600, 18432, 1);

    tot = 1152LL * 448;
    im2col_kernel<<<CEILDIV(tot,256),256,0,stream>>>(d1o, colbuf, 8,128,14,14,
                                                     3,3,2,1,7,7, 448, 1600);
    dim3 g2(128/128, 448/64);
    gemm_wmma_bf16<<<g2,256,0,stream>>>(wd2, colbuf, d2o, b_d2, 128, 448, 1152, 1);

    tot = 1152LL * 128;
    im2col_kernel<<<CEILDIV(tot,256),256,0,stream>>>(d2o, colbuf, 8,128,7,7,
                                                     3,3,2,1,4,4, 128, 448);
    dim3 g3(128/128, 128/64);
    gemm_wmma_bf16<<<g3,256,0,stream>>>(wd3, colbuf, d3o, b_d3, 128, 128, 1152, 1);

    tconv_kernel<<<CEILDIV(8*6*196,256),256,0,stream>>>(d1o, W_t1, b_t1, rpn, 6, 196, 1600, 8, 0);
    tconv_kernel<<<CEILDIV(8*6*49, 256),256,0,stream>>>(d2o, W_t2, b_t2, rpn, 6, 49,  448,  8, 1176);
    tconv_kernel<<<CEILDIV(8*9*16, 256),256,0,stream>>>(d3o, W_t3, b_t3, rpn, 9, 16,  128,  8, 1470);
  }

  // 4) hard NMS (anchors generated on device), writes top_idx / top_prob
  nms_kernel<<<8,256,0,stream>>>(rpn, boxes, out, 9600, 9632);

  // 5) crop + resize parts
  crop_kernel<<<CEILDIV(32*3*PARTSZ*PARTSZ,256),256,0,stream>>>(x, boxes, parts);

  // 6) part backbone (same weights): im2col + WMMA GEMM, reuse conv_out
  {
    long long tot = 3072LL * 1600;
    im2col_kernel<<<CEILDIV(tot,256),256,0,stream>>>(parts, colbuf, 32,3,PARTSZ,PARTSZ,
                                                     32,32,32,0,7,7, 1600, 0);
    dim3 g(2048/128, 1600/64);
    gemm_wmma_bf16<<<g,256,0,stream>>>(wbb, colbuf, conv_out, b_bb, 2048, 1600, 3072, 1);
    mean_pool_kernel<<<CEILDIV(32*2048,256),256,0,stream>>>(conv_out, pfeats, 2048, 1600, 49, 32);
  }

  // 7) heads
  concat_fc_kernel<<<CEILDIV(8*NCLASS,256),256,0,stream>>>(pfeats, feature, W_cat, b_cat,
                                                           out + 1600, 8, NCLASS);
  fc_kernel<<<CEILDIV(32*NCLASS,256),256,0,stream>>>(pfeats, W_part, b_part,
                                                     out + 3200, 32, NCLASS, 2048);
}
